// BidirLSTMLayer_66795331388079
// MI455X (gfx1250) — compile-verified, matrix operand order checked by reading, not run on hardware
//
#include <hip/hip_runtime.h>
#include <hip/hip_bf16.h>

// ---------------------------------------------------------------------------
// Bidirectional LSTM for MI455X (gfx1250), wave32 + WMMA bf16 + async LDS DMA.
//
// Persistent per-direction grids (32 WGs each) step through time; each step
// does the fused [x_t | h_{t-1}] @ [W_ih ; W_hh]^T gate GEMM (K=2048) with
// v_wmma_f32_16x16x32_bf16. A-fragments are staged global->LDS with
// GLOBAL_LOAD_ASYNC_TO_LDS_B128 (ASYNCcnt), double-buffered so the next
// chunk's DMA overlaps the current chunk's WMMAs. Weights live in L2 as
// fragment-packed bf16. A device-scope atomic barrier separates steps.
// ---------------------------------------------------------------------------

#define T_STEPS 512
#define BATCH   64
#define INF_    1024          // input feature size I
#define HID     1024          // hidden size H
#define KTOT    2048          // I + H
#define KC      256           // K chunk staged in LDS (32 KB per buffer)
#define NCH     (KTOT / KC)   // 8 chunks
#define NBLK    32            // workgroups per direction

typedef __attribute__((ext_vector_type(16))) __bf16 v16bf;
typedef __attribute__((ext_vector_type(8)))  float  v8f;
typedef __attribute__((ext_vector_type(4)))  unsigned int u32x4;

union Frag { u32x4 u[2]; v16bf v; };

__device__ __forceinline__ unsigned short f2bf(float f) {
    unsigned int u = __float_as_uint(f);
    u += 0x7FFFu + ((u >> 16) & 1u);       // round-to-nearest-even
    return (unsigned short)(u >> 16);
}

// k position inside a 16x32 (or 32x16) bf16 WMMA fragment for (lane, elem e):
// e=0..7 -> k = klocal+e ; e=8..15 -> k = klocal+16+(e-8); klocal += 8 for
// lanes 16..31.  I.e. per lane: two contiguous 16-byte runs (async-DMA-able).
__device__ __forceinline__ int frag_k(int lane, int e) {
    int r = e >> 1;
    return 2 * r + ((r >= 4) ? 8 : 0) + (e & 1) + ((lane >> 4) * 8);
}

// ---- CDNA5 async global->LDS copy (16B per lane), tracked by ASYNCcnt ----
__device__ __forceinline__ void async_g2l_b128(unsigned int lds_addr,
                                               unsigned long long gaddr) {
    asm volatile("global_load_async_to_lds_b128 %0, %1, off"
                 :: "v"(lds_addr), "v"(gaddr) : "memory");
}
__device__ __forceinline__ void wait_async_le8() {
#if __has_builtin(__builtin_amdgcn_s_wait_asynccnt)
    __builtin_amdgcn_s_wait_asynccnt(8);
#else
    asm volatile("s_wait_asynccnt 0x8" ::: "memory");
#endif
}
__device__ __forceinline__ void wait_async_le0() {
#if __has_builtin(__builtin_amdgcn_s_wait_asynccnt)
    __builtin_amdgcn_s_wait_asynccnt(0);
#else
    asm volatile("s_wait_asynccnt 0x0" ::: "memory");
#endif
}

// ---------------------------------------------------------------------------
// x fp32 -> bf16 (row-major [T*B, I])
// ---------------------------------------------------------------------------
__global__ void convert_x_kernel(const float* __restrict__ x,
                                 unsigned short* __restrict__ xb, int n) {
    for (int i = blockIdx.x * blockDim.x + threadIdx.x; i < n;
         i += gridDim.x * blockDim.x)
        xb[i] = f2bf(x[i]);
}

// ---------------------------------------------------------------------------
// Pack W = [W_ih ; W_hh]^T (K=2048 x N=4096) per direction into fragment-major
// bf16: wp[dir][ntile][kfrag][lane][e]; each lane of a wave reads its 32B
// v16bf slice contiguously from L2.
// ---------------------------------------------------------------------------
__global__ void pack_w_kernel(const float* __restrict__ wih_f,
                              const float* __restrict__ whh_f,
                              const float* __restrict__ wih_b,
                              const float* __restrict__ whh_b,
                              unsigned short* __restrict__ wp) {
    const int per_dir = (KTOT / 32) * (4 * HID / 16) * 512;  // 8388608
    const int n = 2 * per_dir;
    for (int idx = blockIdx.x * blockDim.x + threadIdx.x; idx < n;
         idx += gridDim.x * blockDim.x) {
        int dir  = idx / per_dir;
        int rem  = idx - dir * per_dir;
        int ntile =  rem >> 15;          // / (64*512)
        int kfrag = (rem >> 9) & 63;
        int lane  = (rem >> 4) & 31;
        int e     =  rem & 15;
        int ncol  = ntile * 16 + (lane & 15);          // 0..4095 (gate-major)
        int k     = kfrag * 32 + frag_k(lane, e);      // 0..2047
        const float* wih = dir ? wih_b : wih_f;
        const float* whh = dir ? whh_b : whh_f;
        float v = (k < INF_) ? wih[(size_t)ncol * INF_ + k]
                             : whh[(size_t)ncol * HID + (k - INF_)];
        wp[idx] = f2bf(v);
    }
}

// ---------------------------------------------------------------------------
// Init: sync counters, fused bias, h0 (bf16, parity 0), c0 (fp32)
// ---------------------------------------------------------------------------
__global__ void init_kernel(const float* __restrict__ h0f, const float* __restrict__ c0f,
                            const float* __restrict__ h0b, const float* __restrict__ c0b,
                            const float* __restrict__ bihf, const float* __restrict__ bhhf,
                            const float* __restrict__ bihb, const float* __restrict__ bhhb,
                            unsigned short* __restrict__ hst, float* __restrict__ cst,
                            float* __restrict__ bias, unsigned int* __restrict__ sync) {
    int idx = blockIdx.x * blockDim.x + threadIdx.x;   // 512*256 = 131072
    if (idx < 4) sync[idx] = 0u;
    if (idx < 2 * 4 * HID) {
        int dir = idx >> 12, nn = idx & 4095;
        bias[idx] = dir ? (bihb[nn] + bhhb[nn]) : (bihf[nn] + bhhf[nn]);
    }
    if (idx < 2 * BATCH * HID) {
        int dir = idx >> 16, i = idx & 65535;
        const float* h0 = dir ? h0b : h0f;
        const float* c0 = dir ? c0b : c0f;
        hst[(size_t)(dir * 2 + 0) * (BATCH * HID) + i] = f2bf(h0[i]);
        cst[(size_t)dir * (BATCH * HID) + i] = c0[i];
    }
}

// ---------------------------------------------------------------------------
// Persistent recurrent kernel. grid = (NBLK, 2): blockIdx.y = direction.
// ---------------------------------------------------------------------------
__global__ __launch_bounds__(256, 1)
void lstm_persistent_kernel(const unsigned short* __restrict__ xb,
                            const unsigned short* __restrict__ wp,
                            const float* __restrict__ bias,
                            unsigned short* __restrict__ hst,
                            float* __restrict__ cst,
                            unsigned int* __restrict__ sync,
                            float* __restrict__ out) {
    // two 32 KB A-chunk buffers (fragment order); fp32 gate exchange aliases
    // buffer 0 (dead once the last chunk, which lives in buffer 1, computes).
    __shared__ unsigned short sbuf[2][BATCH * KC];
    float* gbuf = (float*)sbuf[0];               // 8 x [64 x 16] fp32 = 32 KB

    const int tid   = threadIdx.x;
    const int dir   = blockIdx.y;
    const int jb    = blockIdx.x;                 // 32 hidden cols per WG
    const int lane  = tid & 31;
    const int wave  = tid >> 5;                   // 8 waves
    const int gate  = wave & 3;
    const int jslot = wave >> 2;                  // 0..1

    const int per_dir_w = (KTOT / 32) * (4 * HID / 16) * 512;
    const int ntile = gate * (HID / 16) + jb * 2 + jslot;       // 0..255
    const unsigned short* wnt =
        wp + (size_t)dir * per_dir_w + (size_t)ntile * (KTOT / 32) * 512;
    const float* bdir = bias + dir * 4 * HID;
    float* cdir = cst + (size_t)dir * BATCH * HID;
    unsigned int* cnt = sync + dir * 2;
    unsigned int* gen = sync + dir * 2 + 1;

    // per-wave async staging: fragments f = wave*4 .. wave*4+3 of each chunk.
    // fragment f: mtile = f>>3, kfrag = f&7; per lane two 16B runs.
    const int sb0 = ((wave * 4) >> 3) * 16;       // constant pieces hoisted

    for (int s = 0; s < T_STEPS; ++s) {
        const int tstep = dir ? (T_STEPS - 1 - s) : s;
        const int rp = s & 1;           // h parity ping-pong
        const unsigned short* hrd = hst + (size_t)(dir * 2 + rp) * (BATCH * HID);
        unsigned short* hwr = hst + (size_t)(dir * 2 + (rp ^ 1)) * (BATCH * HID);
        const unsigned short* xrow0 = xb + (size_t)tstep * BATCH * INF_;

        v8f acc[4] = {};                // 4 M-tiles x 16 cols, fp32
        (void)sb0;

        // ---- issue async DMA for chunk 0 (x-only) ----
        {
            const unsigned short* src0 = xrow0;          // chunk 0 k-offset 0
#pragma unroll
            for (int i = 0; i < 4; ++i) {
                int f = wave * 4 + i;
                int b = (f >> 3) * 16 + (lane & 15);
                int klocal = (f & 7) * 32 + ((lane >> 4) * 8);
                const unsigned short* g = src0 + (size_t)b * 1024 + klocal;
                unsigned int l0 = (unsigned int)(unsigned long long)
                                  (&sbuf[0][f * 512 + lane * 16]);
                async_g2l_b128(l0,      (unsigned long long)g);
                async_g2l_b128(l0 + 16, (unsigned long long)(g + 16));
            }
        }

        for (int ch = 0; ch < NCH; ++ch) {
            // ---- issue async DMA for chunk ch+1 into the other buffer ----
            if (ch + 1 < NCH) {
                int nc = ch + 1;
                const unsigned short* src0 = (nc < NCH / 2)
                    ? (xrow0 + nc * KC)
                    : (hrd + (nc - NCH / 2) * KC);
                unsigned short* dbuf = sbuf[nc & 1];
#pragma unroll
                for (int i = 0; i < 4; ++i) {
                    int f = wave * 4 + i;
                    int b = (f >> 3) * 16 + (lane & 15);
                    int klocal = (f & 7) * 32 + ((lane >> 4) * 8);
                    const unsigned short* g = src0 + (size_t)b * 1024 + klocal;
                    unsigned int l0 = (unsigned int)(unsigned long long)
                                      (&dbuf[f * 512 + lane * 16]);
                    async_g2l_b128(l0,      (unsigned long long)g);
                    async_g2l_b128(l0 + 16, (unsigned long long)(g + 16));
                }
                wait_async_le8();       // in-order: chunk ch's 8 ops are done
            } else {
                wait_async_le0();
            }
            __syncthreads();            // all waves' DMA visible

            const unsigned short* sA = sbuf[ch & 1];
            // ---- GEMM over this chunk: 8 k-frags x 4 M-tiles ----
#pragma unroll 4
            for (int kf = 0; kf < KC / 32; ++kf) {
                Frag bu;
                const u32x4* wq = (const u32x4*)(
                    wnt + (size_t)(ch * (KC / 32) + kf) * 512 + lane * 16);
                bu.u[0] = wq[0]; bu.u[1] = wq[1];
#pragma unroll
                for (int m = 0; m < 4; ++m) {
                    Frag au;
                    const u32x4* aq =
                        (const u32x4*)(sA + ((m * (KC / 32) + kf) * 512 + lane * 16));
                    au.u[0] = aq[0]; au.u[1] = aq[1];
                    acc[m] = __builtin_amdgcn_wmma_f32_16x16x32_bf16(
                        false, au.v, false, bu.v, (short)0, acc[m], false, false);
                }
            }
            __syncthreads();            // buffer (ch&1) free for reuse
        }

        // ---- exchange fp32 gate tiles through LDS (aliases buffer 0) ----
        {
            float* gb = gbuf + (size_t)(jslot * 4 + gate) * (BATCH * 16);
            int mrow = (lane >> 4) * 8;
            int nn = lane & 15;
#pragma unroll
            for (int m = 0; m < 4; ++m)
#pragma unroll
                for (int r = 0; r < 8; ++r)
                    gb[(m * 16 + r + mrow) * 16 + nn] = acc[m][r];
        }
        __syncthreads();

        // ---- pointwise LSTM cell for this WG's 32 hidden columns ----
#pragma unroll
        for (int i = 0; i < 8; ++i) {
            int e = tid * 8 + i;            // 0..2047
            int b = e >> 5;
            int j = e & 31;
            int js = j >> 4, jj = j & 15;
            int jg = jb * 32 + j;
            float gi = gbuf[((js * 4 + 0) * BATCH + b) * 16 + jj] + bdir[0 * HID + jg];
            float gf = gbuf[((js * 4 + 1) * BATCH + b) * 16 + jj] + bdir[1 * HID + jg];
            float gg = gbuf[((js * 4 + 2) * BATCH + b) * 16 + jj] + bdir[2 * HID + jg];
            float go = gbuf[((js * 4 + 3) * BATCH + b) * 16 + jj] + bdir[3 * HID + jg];
            size_t ci = (size_t)b * HID + jg;
            float c = cdir[ci];
            float ig = 1.0f / (1.0f + __expf(-gi));
            float fg = 1.0f / (1.0f + __expf(-gf));
            float og = 1.0f / (1.0f + __expf(-go));
            float gt = tanhf(gg);
            c = fg * c + ig * gt;
            float h = og * tanhf(c);
            cdir[ci] = c;
            hwr[ci] = f2bf(h);
            out[((size_t)tstep * BATCH + b) * (2 * HID) + dir * HID + jg] = h;
            if (s == T_STEPS - 1) {         // final h/c of this direction
                float* ht = out + (size_t)T_STEPS * BATCH * 2 * HID
                                + (size_t)dir * 2 * BATCH * HID;
                ht[ci] = h;
                ht[BATCH * HID + ci] = c;
            }
        }

        // ---- device-scope barrier across the 32 WGs of this direction ----
        __syncthreads();
        if (tid == 0) {
            __threadfence();
            unsigned int g = __hip_atomic_load(gen, __ATOMIC_RELAXED,
                                               __HIP_MEMORY_SCOPE_AGENT);
            unsigned int a = __hip_atomic_fetch_add(cnt, 1u, __ATOMIC_ACQ_REL,
                                                    __HIP_MEMORY_SCOPE_AGENT);
            if (a == NBLK - 1) {
                __hip_atomic_store(cnt, 0u, __ATOMIC_RELAXED,
                                   __HIP_MEMORY_SCOPE_AGENT);
                __hip_atomic_fetch_add(gen, 1u, __ATOMIC_RELEASE,
                                       __HIP_MEMORY_SCOPE_AGENT);
            } else {
                while (__hip_atomic_load(gen, __ATOMIC_ACQUIRE,
                                         __HIP_MEMORY_SCOPE_AGENT) == g)
                    __builtin_amdgcn_s_sleep(2);
            }
        }
        __syncthreads();
    }
}

// ---------------------------------------------------------------------------
extern "C" void kernel_launch(void* const* d_in, const int* in_sizes, int n_in,
                              void* d_out, int out_size, void* d_ws, size_t ws_size,
                              hipStream_t stream) {
    (void)in_sizes; (void)n_in; (void)out_size; (void)ws_size;
    const float* x    = (const float*)d_in[0];
    const float* h0f  = (const float*)d_in[1];
    const float* c0f  = (const float*)d_in[2];
    const float* h0b  = (const float*)d_in[3];
    const float* c0b  = (const float*)d_in[4];
    const float* wihf = (const float*)d_in[5];
    const float* whhf = (const float*)d_in[6];
    const float* bihf = (const float*)d_in[7];
    const float* bhhf = (const float*)d_in[8];
    const float* wihb = (const float*)d_in[9];
    const float* whhb = (const float*)d_in[10];
    const float* bihb = (const float*)d_in[11];
    const float* bhhb = (const float*)d_in[12];
    float* out = (float*)d_out;

    // workspace carve-out (~102 MB)
    char* ws = (char*)d_ws;
    size_t off = 0;
    auto take = [&](size_t bytes) {
        size_t o = off;
        off = (off + bytes + 255) & ~(size_t)255;
        return o;
    };
    unsigned int*   sync = (unsigned int*)  (ws + take(4 * sizeof(unsigned int)));
    float*          bias = (float*)         (ws + take(2 * 4 * HID * sizeof(float)));
    float*          cst  = (float*)         (ws + take(2 * BATCH * HID * sizeof(float)));
    unsigned short* hst  = (unsigned short*)(ws + take(4 * BATCH * HID * sizeof(unsigned short)));
    unsigned short* xb   = (unsigned short*)(ws + take((size_t)T_STEPS * BATCH * INF_ * sizeof(unsigned short)));
    unsigned short* wpk  = (unsigned short*)(ws + take((size_t)2 * KTOT * 4 * HID * sizeof(unsigned short)));

    const int nx = T_STEPS * BATCH * INF_;
    convert_x_kernel<<<4096, 256, 0, stream>>>(x, xb, nx);
    pack_w_kernel<<<4096, 256, 0, stream>>>(wihf, whhf, wihb, whhb, wpk);
    init_kernel<<<512, 256, 0, stream>>>(h0f, c0f, h0b, c0b,
                                         bihf, bhhf, bihb, bhhb,
                                         hst, cst, bias, sync);
    lstm_persistent_kernel<<<dim3(NBLK, 2), 256, 0, stream>>>(
        xb, wpk, bias, hst, cst, sync, out);
}
